// MultiHeadedAttention_57904749084930
// MI455X (gfx1250) — matrix operand order repaired and checked by reading, not compiled, not run on hardware
//
#include <hip/hip_runtime.h>
#include <hip/hip_bf16.h>
#include <math.h>

// ---------------------------------------------------------------------------
// MHA forward for S=2048, H=2048, 16 heads, hd=128, full-head RoPE.
// All matmuls via v_wmma_f32_16x16x32_bf16 (wave32, CDNA5).
// ---------------------------------------------------------------------------

typedef __bf16 bf16;
typedef __attribute__((ext_vector_type(16))) __bf16 v16bf;
typedef __attribute__((ext_vector_type(8)))  __bf16 v8bf;
typedef __attribute__((ext_vector_type(4)))  __bf16 v4bf;
typedef __attribute__((ext_vector_type(8)))  float  v8f;

#define S_DIM 2048
#define H_DIM 2048
#define NH    16
#define HD    128

// A-matrix fragment (16x32 bf16, K-major source row):
// lanes 0-15 hold K=0..15, lanes 16-31 K=16..31 (caller adds half*16)
// Two adjacent 16B loads, concatenated without per-element movs.
__device__ __forceinline__ v16bf ldfragA(const bf16* p) {
  v8bf lo = *(const v8bf*)(p);
  v8bf hi = *(const v8bf*)(p + 8);
  return __builtin_shufflevector(lo, hi, 0, 1, 2, 3, 4, 5, 6, 7,
                                 8, 9, 10, 11, 12, 13, 14, 15);
}

// B-matrix fragment (32x16 bf16, from K-major B^T row):
// lanes 0-15 hold K=0..15, lanes 16-31 K=16..31 (caller adds half*16)
// 32 contiguous bytes -> single v16bf load (two b128 into adjacent VGPRs).
__device__ __forceinline__ v16bf ldfragB(const bf16* p) {
  return *(const v16bf*)(p);
}

__device__ __forceinline__ v8f wmma_bf16(v16bf a, v16bf b, v8f c) {
  return __builtin_amdgcn_wmma_f32_16x16x32_bf16(
      /*neg_a=*/false, a, /*neg_b=*/false, b,
      /*c_mod=*/(short)0, c, /*reuse_a=*/false, /*reuse_b=*/false);
}

// ---------------------------------------------------------------------------
// Prep kernels
// ---------------------------------------------------------------------------
// Vectorized f32 -> bf16 (4 elements / thread: b128 load, b64 store)
__global__ void cvt_f32_bf16(const float* __restrict__ x, bf16* __restrict__ y, int n4) {
  int i = blockIdx.x * blockDim.x + threadIdx.x;
  if (i >= n4) return;
  float4 v = ((const float4*)x)[i];
  v4bf o;
  o[0] = (bf16)v.x; o[1] = (bf16)v.y; o[2] = (bf16)v.z; o[3] = (bf16)v.w;
  ((v4bf*)y)[i] = o;
}

// LDS-tiled transpose+convert: Wt[n][k] = (bf16) W[k][n]   (H x H)
// block (32,8), 32x32 tile, coalesced on both sides.
__global__ __launch_bounds__(256) void transpose_f32_bf16(
    const float* __restrict__ W, bf16* __restrict__ Wt)
{
  __shared__ bf16 tile[32][33];  // pad to avoid bank conflicts
  const int bx = blockIdx.x * 32;  // n-range on input
  const int by = blockIdx.y * 32;  // k-range on input
  const int tx = threadIdx.x;
  const int ty = threadIdx.y;
#pragma unroll
  for (int yy = 0; yy < 32; yy += 8)
    tile[ty + yy][tx] = (bf16)W[(size_t)(by + ty + yy) * H_DIM + bx + tx];
  __syncthreads();
#pragma unroll
  for (int yy = 0; yy < 32; yy += 8)
    Wt[(size_t)(bx + ty + yy) * H_DIM + by + tx] = tile[tx][ty + yy];
}

// ---------------------------------------------------------------------------
// GEMM: C = A(MxK bf16, row-major) @ Bt^T (Bt is NxK bf16, K-major) + bias
// MODE 0: fp32 row-major out.  MODE 2: bf16 transposed out C[col*M+row].
// Block = 128 threads (4 waves, 2x2), wave tile 64x64, block tile 128x128.
// ---------------------------------------------------------------------------
template <int MODE>
__global__ __launch_bounds__(128) void gemm_bf16_wmma(
    const bf16* __restrict__ A, const bf16* __restrict__ Bt,
    const float* __restrict__ bias, void* __restrict__ Cout,
    int M, int N, int K)
{
  const int lane = threadIdx.x & 31;
  const int wave = threadIdx.x >> 5;
  const int r  = lane & 15;
  const int hf = lane >> 4;
  const int m0 = blockIdx.y * 128 + (wave >> 1) * 64;
  const int n0 = blockIdx.x * 128 + (wave & 1) * 64;

  const bf16* Abase = A  + (size_t)m0 * K;
  const bf16* Bbase = Bt + (size_t)n0 * K;

  v8f acc[4][4] = {};

  for (int k0 = 0; k0 < K; k0 += 32) {
    if (k0 + 32 < K) {  // hint next k-slab toward the caches
      __builtin_prefetch((const void*)(Abase + (size_t)r * K + k0 + 32), 0, 1);
      __builtin_prefetch((const void*)(Bbase + (size_t)r * K + k0 + 32), 0, 1);
    }
    v16bf af[4], bfr[4];
#pragma unroll
    for (int i = 0; i < 4; ++i)
      af[i] = ldfragA(Abase + (size_t)(16 * i + r) * K + k0 + hf * 16);
#pragma unroll
    for (int j = 0; j < 4; ++j)
      bfr[j] = ldfragB(Bbase + (size_t)(16 * j + r) * K + k0 + hf * 16);
#pragma unroll
    for (int i = 0; i < 4; ++i)
#pragma unroll
      for (int j = 0; j < 4; ++j)
        acc[i][j] = wmma_bf16(af[i], bfr[j], acc[i][j]);
  }

#pragma unroll
  for (int j = 0; j < 4; ++j) {
    const int col = n0 + 16 * j + r;
    const float bval = bias ? bias[col] : 0.0f;
#pragma unroll
    for (int i = 0; i < 4; ++i) {
#pragma unroll
      for (int e = 0; e < 8; ++e) {
        const int row = m0 + 16 * i + 8 * hf + e;
        const float v = acc[i][j][e] + bval;
        if (MODE == 0)
          ((float*)Cout)[(size_t)row * N + col] = v;
        else
          ((bf16*)Cout)[(size_t)col * M + row] = (bf16)v;  // transposed store
      }
    }
  }
}

// ---------------------------------------------------------------------------
// RoPE (full head dim 128): rotate-half with on-the-fly cos/sin, fp32 -> bf16
// ---------------------------------------------------------------------------
__global__ void rope_bf16(const float* __restrict__ Qf, const float* __restrict__ Kf,
                          bf16* __restrict__ Qh, bf16* __restrict__ Kh)
{
  int idx = blockIdx.x * blockDim.x + threadIdx.x;
  if (idx >= S_DIM * NH * (HD / 2)) return;
  const int i = idx & 63;
  const int h = (idx >> 6) & (NH - 1);
  const int s = idx >> 10;
  // inv_freq = 10000^{-2i/128} = exp(-(2i/128) * ln 10000)
  const float inv_freq = __expf(-(float)(2 * i) * (9.210340371976184f / 128.0f));
  const float ang = (float)s * inv_freq;
  float c, sn;
  __sincosf(ang, &c, &sn);
  const size_t base = (size_t)s * H_DIM + (size_t)h * HD;
  const float q1 = Qf[base + i], q2 = Qf[base + 64 + i];
  Qh[base + i]      = (bf16)(q1 * c - q2 * sn);
  Qh[base + 64 + i] = (bf16)(q2 * c + q1 * sn);
  const float k1 = Kf[base + i], k2 = Kf[base + 64 + i];
  Kh[base + i]      = (bf16)(k1 * c - k2 * sn);
  Kh[base + 64 + i] = (bf16)(k2 * c + k1 * sn);
}

// ---------------------------------------------------------------------------
// Flash attention: one wave per (32 queries, head). Online softmax in the
// WMMA C-fragment layout; P staged through LDS to become an A-fragment.
//   Qh, Kh: (S, H) bf16 (roped).  Vt: (NH*HD, S) bf16.  attn: (S, H) bf16.
// ---------------------------------------------------------------------------
__global__ __launch_bounds__(32) void flash_attn_wmma(
    const bf16* __restrict__ Qh, const bf16* __restrict__ Kh,
    const bf16* __restrict__ Vt, bf16* __restrict__ attn)
{
  __shared__ bf16 Pt[32 * 64];
  const int lane = threadIdx.x & 31;
  const int r  = lane & 15;
  const int hf = lane >> 4;
  const int h  = blockIdx.y;
  const int q0 = blockIdx.x * 32;
  const int hb = h * HD;
  const float scale = 0.08838834764831845f;  // 1/sqrt(128)

  // Q tile 32x128 resident in VGPRs: 2 m-frags x 4 k-steps
  v16bf qf[2][4];
#pragma unroll
  for (int mi = 0; mi < 2; ++mi)
#pragma unroll
    for (int kk = 0; kk < 4; ++kk)
      qf[mi][kk] = ldfragA(Qh + (size_t)(q0 + 16 * mi + r) * H_DIM + hb + kk * 32 + hf * 16);

  v8f o[2][8] = {};              // 32x128 fp32 output accumulator (C layout)
  float mrow[2][8], lrow[2][8];  // per-row running max / sum (row = 16*mi + 8*hf + e)
#pragma unroll
  for (int mi = 0; mi < 2; ++mi)
#pragma unroll
    for (int e = 0; e < 8; ++e) { mrow[mi][e] = -INFINITY; lrow[mi][e] = 0.0f; }

  for (int kb = 0; kb < S_DIM; kb += 64) {
    // ---- scores S = Q K^T for 64 keys (4 n-frags) ----
    v8f sc[2][4] = {};
#pragma unroll
    for (int j = 0; j < 4; ++j) {
      v16bf kf[4];
#pragma unroll
      for (int kk = 0; kk < 4; ++kk)
        kf[kk] = ldfragB(Kh + (size_t)(kb + 16 * j + r) * H_DIM + hb + kk * 32 + hf * 16);
#pragma unroll
      for (int mi = 0; mi < 2; ++mi)
#pragma unroll
        for (int kk = 0; kk < 4; ++kk)
          sc[mi][j] = wmma_bf16(qf[mi][kk], kf[kk], sc[mi][j]);
    }

    // ---- online softmax per row (reduce across the 16-lane half-group) ----
#pragma unroll
    for (int mi = 0; mi < 2; ++mi) {
#pragma unroll
      for (int e = 0; e < 8; ++e) {
        float rm = -INFINITY;
#pragma unroll
        for (int j = 0; j < 4; ++j) {
          sc[mi][j][e] *= scale;
          rm = fmaxf(rm, sc[mi][j][e]);
        }
#pragma unroll
        for (int off = 8; off >= 1; off >>= 1)
          rm = fmaxf(rm, __shfl_xor(rm, off, 32));
        const float mnew = fmaxf(mrow[mi][e], rm);
        const float corr = __expf(mrow[mi][e] - mnew);
        mrow[mi][e] = mnew;
        float rsum = 0.0f;
#pragma unroll
        for (int j = 0; j < 4; ++j) {
          const float p = __expf(sc[mi][j][e] - mnew);
          sc[mi][j][e] = p;
          rsum += p;
        }
#pragma unroll
        for (int off = 8; off >= 1; off >>= 1)
          rsum += __shfl_xor(rsum, off, 32);
        lrow[mi][e] = lrow[mi][e] * corr + rsum;
#pragma unroll
        for (int n = 0; n < 8; ++n)
          o[mi][n][e] *= corr;
      }
    }

    // ---- C-layout P -> LDS (row-major 32x64 bf16) ----
#pragma unroll
    for (int mi = 0; mi < 2; ++mi)
#pragma unroll
      for (int j = 0; j < 4; ++j)
#pragma unroll
        for (int e = 0; e < 8; ++e)
          Pt[(16 * mi + 8 * hf + e) * 64 + 16 * j + r] = (bf16)sc[mi][j][e];

    // ---- re-load P as A-fragments (single-wave block: DS ops in-order) ----
    v16bf pa[2][2];
#pragma unroll
    for (int mi = 0; mi < 2; ++mi)
#pragma unroll
      for (int kk = 0; kk < 2; ++kk)
        pa[mi][kk] = ldfragA(&Pt[(16 * mi + r) * 64 + kk * 32 + hf * 16]);

    // ---- O += P @ V (Vt is K-contiguous: B^T rows are head dims) ----
#pragma unroll
    for (int n = 0; n < 8; ++n) {
      v16bf vb[2];
#pragma unroll
      for (int kk = 0; kk < 2; ++kk)
        vb[kk] = ldfragB(Vt + (size_t)(hb + 16 * n + r) * S_DIM + kb + kk * 32 + hf * 16);
#pragma unroll
      for (int mi = 0; mi < 2; ++mi)
#pragma unroll
        for (int kk = 0; kk < 2; ++kk)
          o[mi][n] = wmma_bf16(pa[mi][kk], vb[kk], o[mi][n]);
    }
  }

  // ---- epilogue: O / l, store bf16 attn (S, H) ----
#pragma unroll
  for (int mi = 0; mi < 2; ++mi)
#pragma unroll
    for (int e = 0; e < 8; ++e) {
      const float inv = 1.0f / lrow[mi][e];
      const int row = q0 + 16 * mi + 8 * hf + e;
#pragma unroll
      for (int n = 0; n < 8; ++n)
        attn[(size_t)row * H_DIM + hb + 16 * n + r] = (bf16)(o[mi][n][e] * inv);
    }
}

// ---------------------------------------------------------------------------
// Launch
// ---------------------------------------------------------------------------
extern "C" void kernel_launch(void* const* d_in, const int* in_sizes, int n_in,
                              void* d_out, int out_size, void* d_ws, size_t ws_size,
                              hipStream_t stream)
{
  (void)in_sizes; (void)n_in; (void)out_size; (void)ws_size;
  const float* X  = (const float*)d_in[0];
  const float* wq = (const float*)d_in[1];
  const float* bq = (const float*)d_in[2];
  const float* wk = (const float*)d_in[3];
  const float* bk = (const float*)d_in[4];
  const float* wv = (const float*)d_in[5];
  const float* bv = (const float*)d_in[6];
  const float* wo = (const float*)d_in[7];
  const float* bo = (const float*)d_in[8];
  float* out = (float*)d_out;

  // Workspace layout (96 MB total):
  char* ws = (char*)d_ws;
  const size_t MB = 1024 * 1024;
  bf16*  Xh  = (bf16*) (ws +  0 * MB);  //  8 MB  X in bf16
  bf16*  WqT = (bf16*) (ws +  8 * MB);  //  8 MB  wq^T bf16
  bf16*  WkT = (bf16*) (ws + 16 * MB);  //  8 MB
  bf16*  WvT = (bf16*) (ws + 24 * MB);  //  8 MB
  bf16*  WoT = (bf16*) (ws + 32 * MB);  //  8 MB
  float* Qf  = (float*)(ws + 40 * MB);  // 16 MB  Q fp32 (pre-rope)
  float* Kf  = (float*)(ws + 56 * MB);  // 16 MB  K fp32 (pre-rope)
  bf16*  Qh  = (bf16*) (ws + 72 * MB);  //  8 MB  roped Q bf16
  bf16*  Kh  = (bf16*) (ws + 80 * MB);  //  8 MB  roped K bf16
  bf16*  Vt  = (bf16*) (ws + 88 * MB);  //  8 MB  V^T per-head (NH*HD, S)
  bf16*  At  = (bf16*) (ws + 40 * MB);  //  8 MB  attn bf16 (reuses dead Qf slot)

  const int NE = H_DIM * H_DIM;  // 4M elements
  cvt_f32_bf16<<<(NE / 4) / 256, 256, 0, stream>>>(X, Xh, NE / 4);

  dim3 tgrid(H_DIM / 32, H_DIM / 32), tblk(32, 8);
  transpose_f32_bf16<<<tgrid, tblk, 0, stream>>>(wq, WqT);
  transpose_f32_bf16<<<tgrid, tblk, 0, stream>>>(wk, WkT);
  transpose_f32_bf16<<<tgrid, tblk, 0, stream>>>(wv, WvT);
  transpose_f32_bf16<<<tgrid, tblk, 0, stream>>>(wo, WoT);

  dim3 ggrid(H_DIM / 128, S_DIM / 128);
  gemm_bf16_wmma<0><<<ggrid, 128, 0, stream>>>(Xh, WqT, bq, Qf, S_DIM, H_DIM, H_DIM);
  gemm_bf16_wmma<0><<<ggrid, 128, 0, stream>>>(Xh, WkT, bk, Kf, S_DIM, H_DIM, H_DIM);
  gemm_bf16_wmma<2><<<ggrid, 128, 0, stream>>>(Xh, WvT, bv, Vt, S_DIM, H_DIM, H_DIM);

  rope_bf16<<<(S_DIM * NH * (HD / 2)) / 256, 256, 0, stream>>>(Qf, Kf, Qh, Kh);

  dim3 fgrid(S_DIM / 32, NH);
  flash_attn_wmma<<<fgrid, 32, 0, stream>>>(Qh, Kh, Vt, At);

  gemm_bf16_wmma<0><<<ggrid, 128, 0, stream>>>(At, WoT, bo, out, S_DIM, H_DIM, H_DIM);
}
